// wide_deep_model_76012331204803
// MI455X (gfx1250) — matrix operand order repaired and checked by reading, not compiled, not run on hardware
//
#include <hip/hip_runtime.h>
#include <hip/hip_bf16.h>
#include <math.h>

// ---- problem constants (match reference) ----
#define B_SZ   32768
#define T_FEAT 26
#define V_SZ   50000
#define D_EMB  128
#define NN_F   13
#define H1     256
#define H2     128
#define KPAD   3360   // 32 (padded numerical) + 26*128 embeddings; 105 chunks of 32

typedef _Float16 __attribute__((ext_vector_type(16))) v16h;
typedef _Float16 __attribute__((ext_vector_type(8)))  v8h;
typedef float    __attribute__((ext_vector_type(8)))  v8f;
typedef float    __attribute__((ext_vector_type(4)))  v4f;

// ---------------------------------------------------------------------------
// prep: W1 [3341,256] f32 row-major -> W1t [256][KPAD] f16 (N-major, K padded)
// original row for padded col k: k<13 -> k ; 13<=k<32 -> zero ; k>=32 -> 13+(k-32)
// ---------------------------------------------------------------------------
__global__ __launch_bounds__(256) void prep_w1_kernel(const float* __restrict__ W1,
                                                      _Float16* __restrict__ W1t) {
  int i = blockIdx.x * 256 + threadIdx.x;          // i < 256*KPAD
  int k = i % KPAD;
  int n = i / KPAD;
  float v = 0.0f;
  if (k < NN_F)       v = W1[k * H1 + n];
  else if (k >= 32)   v = W1[(NN_F + (k - 32)) * H1 + n];
  W1t[i] = (_Float16)v;
}

// W2 [256,128] f32 -> W2t [128][256] f16 (N-major)
__global__ __launch_bounds__(256) void prep_w2_kernel(const float* __restrict__ W2,
                                                      _Float16* __restrict__ W2t) {
  int i = blockIdx.x * 256 + threadIdx.x;          // i < 128*256
  int k = i % H1;
  int n = i / H1;
  W2t[i] = (_Float16)W2[k * H2 + n];
}

// wide[b] = relu(numerical[b,:] . w_wide + b_wide)
__global__ __launch_bounds__(256) void wide_kernel(const float* __restrict__ num,
                                                   const float* __restrict__ w,
                                                   const float* __restrict__ bw,
                                                   float* __restrict__ wide) {
  int b = blockIdx.x * 256 + threadIdx.x;
  float s = bw[0];
#pragma unroll
  for (int k = 0; k < NN_F; ++k) s += num[b * NN_F + k] * w[k];
  wide[b] = fmaxf(s, 0.0f);
}

// ---------------------------------------------------------------------------
// Fused gather + GEMM1 + cat_sum.
//  grid = 256 blocks (M tiles of 128), block = 256 threads = 8 waves.
//  wave w owns rows [m0, m0+16), all 16 N-subtiles of H1=256.
//  A-frag layout (16-bit A 16x32): lane m=lane&15, hi=lane>>4,
//    element e -> K = e + 8*hi + 8*(e>>3)
//  B-frag layout (32x16): lane n=lane&15, element e -> K as for A (chunk 0: K = e + 16*hi in A and B)
//  C-frag layout (16x16 f32): lane n=lane&15, VGPR r -> M = r + 8*hi
// ---------------------------------------------------------------------------
__global__ __launch_bounds__(256) void gemm1_fused_kernel(
    const float* __restrict__ numerical,        // [B,13]
    const int* __restrict__ cat,                // [T,B]
    const float* __restrict__ tables,           // [T,V,D]
    const _Float16* __restrict__ W1t,           // [256][KPAD]
    const float* __restrict__ b1,               // [256]
    _Float16* __restrict__ X1,                  // [B,256] out (post-relu, f16)
    float* __restrict__ cat_sum)                // [B,128] out
{
  const int lane = threadIdx.x & 31;
  const int wave = threadIdx.x >> 5;
  const int nl   = lane & 15;
  const int hi   = lane >> 4;
  const int m0   = blockIdx.x * 128 + wave * 16;
  const int m    = m0 + nl;                     // this lane's A-matrix row

  v8f acc[16];
#pragma unroll
  for (int i = 0; i < 16; ++i) { v8f z = {0.f,0.f,0.f,0.f,0.f,0.f,0.f,0.f}; acc[i] = z; }
  float cs[64];
#pragma unroll
  for (int i = 0; i < 64; ++i) cs[i] = 0.0f;

  // ---- K-chunk 0: numerical features (padded to 32 cols) ----
  {
    v16h a;
#pragma unroll
    for (int e = 0; e < 16; ++e) {
      const int k = e + 16 * hi;
      const float v = (k < NN_F) ? numerical[m * NN_F + k] : 0.0f;
      a[e] = (_Float16)v;
    }
#pragma unroll
    for (int nt = 0; nt < 16; ++nt) {
      const v16h bf = *(const v16h*)(W1t + (size_t)(nt * 16 + nl) * KPAD + 16 * hi);
      acc[nt] = __builtin_amdgcn_wmma_f32_16x16x32_f16(false, a, false, bf,
                                                       (short)0, acc[nt], false, false);
    }
  }

  // ---- embedding K-chunks: 26 tables x 4 chunks of 32 ----
  for (int t = 0; t < T_FEAT; ++t) {
    const int idx = cat[t * B_SZ + m];
    const float* tab = tables + ((size_t)t * V_SZ + (size_t)idx) * D_EMB;
#pragma unroll
    for (int c = 0; c < 4; ++c) {
      const float* p = tab + c * 32 + 8 * hi;
      const v4f r0 = *(const v4f*)(p);          // K-in-chunk = 8hi + 0..3   (e=0..3)
      const v4f r1 = *(const v4f*)(p + 4);      //            = 8hi + 4..7   (e=4..7)
      const v4f r2 = *(const v4f*)(p + 16);     //            = 16+8hi+0..3  (e=8..11)
      const v4f r3 = *(const v4f*)(p + 20);     //            = 16+8hi+4..7  (e=12..15)
      // fused cat_sum accumulation (f32, each element read from HBM exactly once)
#pragma unroll
      for (int j = 0; j < 4; ++j) {
        cs[c * 16 + 0  + j] += r0[j];
        cs[c * 16 + 4  + j] += r1[j];
        cs[c * 16 + 8  + j] += r2[j];
        cs[c * 16 + 12 + j] += r3[j];
      }
      v16h a;
#pragma unroll
      for (int j = 0; j < 4; ++j) {
        a[0  + j] = (_Float16)r0[j];
        a[4  + j] = (_Float16)r1[j];
        a[8  + j] = (_Float16)r2[j];
        a[12 + j] = (_Float16)r3[j];
      }
      const int kk = 32 + (t * 4 + c) * 32;
#pragma unroll
      for (int nt = 0; nt < 16; ++nt) {
        const _Float16* pb = W1t + (size_t)(nt * 16 + nl) * KPAD + kk + 8 * hi;
        const v8h bq0 = *(const v8h*)(pb);
        const v8h bq1 = *(const v8h*)(pb + 16);
        v16h bf;
#pragma unroll
        for (int j = 0; j < 8; ++j) { bf[j] = bq0[j]; bf[8 + j] = bq1[j]; }
        acc[nt] = __builtin_amdgcn_wmma_f32_16x16x32_f16(false, a, false, bf,
                                                         (short)0, acc[nt], false, false);
      }
    }
  }

  // ---- epilogue: cat_sum writes (lane's 64 d-values; hi halves are disjoint) ----
#pragma unroll
  for (int c = 0; c < 4; ++c) {
    float* d0 = cat_sum + (size_t)m * D_EMB + c * 32 + 8 * hi;       // e=0..7
    float* d1 = cat_sum + (size_t)m * D_EMB + c * 32 + 16 + 8 * hi;  // e=8..15
    v4f o0 = {cs[c*16+0], cs[c*16+1], cs[c*16+2],  cs[c*16+3]};
    v4f o1 = {cs[c*16+4], cs[c*16+5], cs[c*16+6],  cs[c*16+7]};
    v4f o2 = {cs[c*16+8], cs[c*16+9], cs[c*16+10], cs[c*16+11]};
    v4f o3 = {cs[c*16+12],cs[c*16+13],cs[c*16+14], cs[c*16+15]};
    *(v4f*)(d0)     = o0;
    *(v4f*)(d0 + 4) = o1;
    *(v4f*)(d1)     = o2;
    *(v4f*)(d1 + 4) = o3;
  }

  // ---- epilogue: bias + relu + f16 store of X1 ----
#pragma unroll
  for (int nt = 0; nt < 16; ++nt) {
    const int n = nt * 16 + nl;
    const float bias = b1[n];
#pragma unroll
    for (int r = 0; r < 8; ++r) {
      float v = acc[nt][r] + bias;
      v = v > 0.0f ? v : 0.0f;
      X1[(size_t)(m0 + hi * 8 + r) * H1 + n] = (_Float16)v;
    }
  }
}

// ---------------------------------------------------------------------------
// GEMM2: X1[B,256] f16 x W2t[128][256] f16 -> X2[B,128] f16 (relu+bias)
// grid = 256 (M tiles of 128), block = 256 (8 waves, 1 M-subtile x 8 N-subtiles)
// ---------------------------------------------------------------------------
__global__ __launch_bounds__(256) void gemm2_kernel(const _Float16* __restrict__ X1,
                                                    const _Float16* __restrict__ W2t,
                                                    const float* __restrict__ b2,
                                                    _Float16* __restrict__ X2) {
  const int lane = threadIdx.x & 31;
  const int wave = threadIdx.x >> 5;
  const int nl   = lane & 15;
  const int hi   = lane >> 4;
  const int m0   = blockIdx.x * 128 + wave * 16;
  const int m    = m0 + nl;

  v8f acc[8];
#pragma unroll
  for (int i = 0; i < 8; ++i) { v8f z = {0.f,0.f,0.f,0.f,0.f,0.f,0.f,0.f}; acc[i] = z; }

#pragma unroll
  for (int ks = 0; ks < 8; ++ks) {
    const int kk = ks * 32;
    const _Float16* pa = X1 + (size_t)m * H1 + kk + 16 * hi;
    const v8h a0 = *(const v8h*)(pa);       // e=0..7  : K = kk + 16hi + e
    const v8h a1 = *(const v8h*)(pa + 8);   // e=8..15 : K = kk + 16hi + e
    v16h a;
#pragma unroll
    for (int j = 0; j < 8; ++j) { a[j] = a0[j]; a[8 + j] = a1[j]; }
#pragma unroll
    for (int nt = 0; nt < 8; ++nt) {
      const v16h bf = *(const v16h*)(W2t + (size_t)(nt * 16 + nl) * H1 + kk + 16 * hi);
      acc[nt] = __builtin_amdgcn_wmma_f32_16x16x32_f16(false, a, false, bf,
                                                       (short)0, acc[nt], false, false);
    }
  }

#pragma unroll
  for (int nt = 0; nt < 8; ++nt) {
    const int n = nt * 16 + nl;
    const float bias = b2[n];
#pragma unroll
    for (int r = 0; r < 8; ++r) {
      float v = acc[nt][r] + bias;
      v = v > 0.0f ? v : 0.0f;
      X2[(size_t)(m0 + hi * 8 + r) * H2 + n] = (_Float16)v;
    }
  }
}

// ---------------------------------------------------------------------------
// Final: dnn[b] = X2[b,:].W3 + b3 ; out[b,d] = sigmoid(dnn + wide[b] + cat_sum[b,d])
// one wave per row b; grid = B/8, block = 256 (8 waves)
// ---------------------------------------------------------------------------
__global__ __launch_bounds__(256) void final_kernel(const _Float16* __restrict__ X2,
                                                    const float* __restrict__ W3,
                                                    const float* __restrict__ b3,
                                                    const float* __restrict__ cat_sum,
                                                    const float* __restrict__ wide,
                                                    float* __restrict__ out) {
  const int lane = threadIdx.x & 31;
  const int wave = threadIdx.x >> 5;
  const int b = blockIdx.x * 8 + wave;
  float s = 0.0f;
#pragma unroll
  for (int j = 0; j < 4; ++j) {
    const int d = lane + j * 32;
    s += (float)X2[(size_t)b * H2 + d] * W3[d];
  }
#pragma unroll
  for (int off = 16; off > 0; off >>= 1) s += __shfl_xor(s, off, 32);
  const float dnn = s + b3[0];
  const float wb  = wide[b];
#pragma unroll
  for (int j = 0; j < 4; ++j) {
    const int d = lane + j * 32;
    const float x = dnn + wb + cat_sum[(size_t)b * H2 + d];
    out[(size_t)b * H2 + d] = 1.0f / (1.0f + __expf(-x));
  }
}

// ---------------------------------------------------------------------------
extern "C" void kernel_launch(void* const* d_in, const int* in_sizes, int n_in,
                              void* d_out, int out_size, void* d_ws, size_t ws_size,
                              hipStream_t stream) {
  (void)in_sizes; (void)n_in; (void)out_size; (void)ws_size;
  const float* numerical = (const float*)d_in[0];
  const int*   cat       = (const int*)  d_in[1];
  const float* tables    = (const float*)d_in[2];
  const float* w_wide    = (const float*)d_in[3];
  const float* b_wide    = (const float*)d_in[4];
  const float* W1        = (const float*)d_in[5];
  const float* b1        = (const float*)d_in[6];
  const float* W2        = (const float*)d_in[7];
  const float* b2        = (const float*)d_in[8];
  const float* W3        = (const float*)d_in[9];
  const float* b3        = (const float*)d_in[10];
  float* out = (float*)d_out;

  // workspace carve-up (~41.9 MB total)
  char* ws = (char*)d_ws;
  _Float16* W1t     = (_Float16*)(ws);                       // 256*3360*2  = 1,720,320
  _Float16* W2t     = (_Float16*)(ws + 1720320);             // 128*256*2   = 65,536
  _Float16* X1      = (_Float16*)(ws + 1785856);             // B*256*2     = 16,777,216
  _Float16* X2      = (_Float16*)(ws + 18563072);            // B*128*2     = 8,388,608
  float*    cat_sum = (float*)   (ws + 26951680);            // B*128*4     = 16,777,216
  float*    wide    = (float*)   (ws + 43728896);            // B*4         = 131,072

  prep_w1_kernel<<<KPAD, 256, 0, stream>>>(W1, W1t);                       // 256*KPAD/256
  prep_w2_kernel<<<H2, 256, 0, stream>>>(W2, W2t);                         // 128*256/256
  wide_kernel<<<B_SZ / 256, 256, 0, stream>>>(numerical, w_wide, b_wide, wide);
  gemm1_fused_kernel<<<B_SZ / 128, 256, 0, stream>>>(numerical, cat, tables,
                                                     W1t, b1, X1, cat_sum);
  gemm2_kernel<<<B_SZ / 128, 256, 0, stream>>>(X1, W2t, b2, X2);
  final_kernel<<<B_SZ / 8, 256, 0, stream>>>(X2, W3, b3, cat_sum, wide, out);
}
